// TKGOnlyModel_61933428408553
// MI455X (gfx1250) — matrix operand order repaired and checked by reading, not compiled, not run on hardware
//
#include <hip/hip_runtime.h>
#include <math.h>

// ---------------------------------------------------------------------------
// GAT (2-layer) + MLP head for MI455X (gfx1250, wave32, WMMA).
// GEMMs run on v_wmma_f32_16x16x32_bf16 (f32 accumulate); attention
// segment-softmax uses ordered-uint atomicMax + f32 atomicAdd. The 192MB L2
// keeps h/acc resident during the edge scatter phase.
// ---------------------------------------------------------------------------

typedef __attribute__((ext_vector_type(16))) __bf16 bf16x16;
typedef __attribute__((ext_vector_type(8)))  float  floatx8;

#define LRELU_SLOPE 0.2f

// Branchless round-to-nearest-even f32 -> bf16 (inputs are finite activations;
// no NaN special case so no EXEC-mask branching in the GEMM hot loop).
__device__ __forceinline__ unsigned short f32_to_bf16_bits(float f) {
  unsigned u = __float_as_uint(f);
  u += 0x7fffu + ((u >> 16) & 1u);
  return (unsigned short)(u >> 16);
}
// Pack two floats into one dword of two bf16 (lo in [15:0], hi in [31:16]).
__device__ __forceinline__ unsigned pack_bf16x2(float lo, float hi) {
  unsigned ul = __float_as_uint(lo);
  unsigned uh = __float_as_uint(hi);
  ul += 0x7fffu + ((ul >> 16) & 1u);
  uh += 0x7fffu + ((uh >> 16) & 1u);
  return (ul >> 16) | (uh & 0xffff0000u);
}
// order-preserving float <-> uint key (for atomicMax-based segment max)
__device__ __forceinline__ unsigned f32_key(float f) {
  unsigned u = __float_as_uint(f);
  return (u & 0x80000000u) ? ~u : (u | 0x80000000u);
}
__device__ __forceinline__ float key_f32(unsigned k) {
  return __uint_as_float((k & 0x80000000u) ? (k ^ 0x80000000u) : ~k);
}

// ---------------------------------------------------------------------------
// Pre-swizzle W (f32 [K,Nout] row-major) into WMMA B fragments (bf16).
// Fragment order: [kt][nt][lane][e], lane<16 -> col=lane, K 0..7,16..23 of 32-group;
// lane>=16 -> col=lane-16, K 8..15,24..31 (same K order as the A fragment).
// ---------------------------------------------------------------------------
__global__ void gat_pack_w(const float* __restrict__ W, unsigned short* __restrict__ Bp,
                           int K, int Nout) {
  int NT = Nout >> 4;
  int total = (K >> 5) * NT * 32 * 16;
  int idx = blockIdx.x * blockDim.x + threadIdx.x;
  if (idx >= total) return;
  int e    = idx & 15;
  int lane = (idx >> 4) & 31;
  int tile = idx >> 9;          // kt*NT + nt
  int nt = tile % NT;
  int kt = tile / NT;
  int col = nt * 16 + (lane & 15);
  int k   = kt * 32 + ((lane >> 4) << 3) + (e & 7) + ((e >> 3) << 4);
  Bp[idx] = f32_to_bf16_bits(W[(size_t)k * Nout + col]);
}

// ---------------------------------------------------------------------------
// C = A(f32 [M,128]) x Bpacked(bf16)  ->  C f32 [M, NT*16]
// 256 threads = 8 waves; each wave: 16 rows x NT*16 cols, K-loop 4 x K32.
// A fragment (16-bit A 16x32 layout): lanes<16 hold K kb..kb+7 and kb+16..kb+23
// with kb=0; lanes>=16 the same with kb=8. Element pairs pack into dwords.
// ---------------------------------------------------------------------------
template <int NT>
__global__ __launch_bounds__(256) void gat_gemm_wmma(
    const float* __restrict__ A, const bf16x16* __restrict__ Bp,
    float* __restrict__ C, int M) {
  constexpr int K = 128;
  constexpr int Nout = NT * 16;
  const int lane = threadIdx.x & 31;
  const int wave = threadIdx.x >> 5;
  const int row0 = blockIdx.x * 128 + wave * 16;

  int arow = row0 + (lane & 15);
  if (arow >= M) arow = M - 1;                 // clamp loads; stores are guarded
  const float* arp = A + (size_t)arow * K;
  const int kb_lane = (lane >> 4) << 3;        // lanes>=16 start at K offset 8

  floatx8 acc[NT];
#pragma unroll
  for (int i = 0; i < NT; ++i)
#pragma unroll
    for (int j = 0; j < 8; ++j) acc[i][j] = 0.0f;

#pragma unroll
  for (int kt = 0; kt < K / 32; ++kt) {
    const int kbase = kt * 32 + kb_lane;
    float4 v0 = *(const float4*)(arp + kbase);
    float4 v1 = *(const float4*)(arp + kbase + 4);
    float4 v2 = *(const float4*)(arp + kbase + 16);
    float4 v3 = *(const float4*)(arp + kbase + 20);
    union { unsigned u[8]; bf16x16 v; } au;
    au.u[0] = pack_bf16x2(v0.x, v0.y);
    au.u[1] = pack_bf16x2(v0.z, v0.w);
    au.u[2] = pack_bf16x2(v1.x, v1.y);
    au.u[3] = pack_bf16x2(v1.z, v1.w);
    au.u[4] = pack_bf16x2(v2.x, v2.y);
    au.u[5] = pack_bf16x2(v2.z, v2.w);
    au.u[6] = pack_bf16x2(v3.x, v3.y);
    au.u[7] = pack_bf16x2(v3.z, v3.w);
    bf16x16 a = au.v;
#pragma unroll
    for (int nt = 0; nt < NT; ++nt) {
      bf16x16 b = Bp[(kt * NT + nt) * 32 + lane];
      acc[nt] = __builtin_amdgcn_wmma_f32_16x16x32_bf16(
          false, a, false, b, (short)0, acc[nt], false, false);
    }
  }

  const int srow = row0 + ((lane >> 4) << 3);  // lanes>=16 hold rows M=8..15
  const int col  = lane & 15;
#pragma unroll
  for (int nt = 0; nt < NT; ++nt)
#pragma unroll
    for (int r = 0; r < 8; ++r) {
      int row = srow + r;
      if (row < M) C[(size_t)row * Nout + nt * 16 + col] = acc[nt][r];
    }
}

// ---------------------------------------------------------------------------
// alpha_s[n,h] = <h[n,h,:], a_src[h,:]> ; same for alpha_d
// ---------------------------------------------------------------------------
__global__ void gat_attn_alpha(const float* __restrict__ h,
                               const float* __restrict__ a_src,
                               const float* __restrict__ a_dst,
                               float* __restrict__ as, float* __restrict__ ad,
                               int N, int H, int F) {
  int idx = blockIdx.x * blockDim.x + threadIdx.x;
  if (idx >= N * H) return;
  int n = idx / H, hh = idx - n * H;
  const float* hp = h + ((size_t)n * H + hh) * F;
  const float* sp = a_src + hh * F;
  const float* dp = a_dst + hh * F;
  float ss = 0.f, dd = 0.f;
  for (int f = 0; f < F; f += 4) {
    float4 hv = *(const float4*)(hp + f);
    float4 sv = *(const float4*)(sp + f);
    float4 dv = *(const float4*)(dp + f);
    ss += hv.x * sv.x + hv.y * sv.y + hv.z * sv.z + hv.w * sv.w;
    dd += hv.x * dv.x + hv.y * dv.y + hv.z * dv.z + hv.w * dv.w;
  }
  as[idx] = ss;
  ad[idx] = dd;
}

// ---------------------------------------------------------------------------
// Pass A: e = leaky_relu(as[src]+ad[dst]); store e; atomicMax key into mkey[dst]
// Edges >= E are the appended self-loops (src=dst=e-E).
// ---------------------------------------------------------------------------
__global__ void gat_edge_logit_max(const int* __restrict__ src, const int* __restrict__ dst,
                                   const float* __restrict__ as, const float* __restrict__ ad,
                                   float* __restrict__ ebuf, unsigned* __restrict__ mkey,
                                   int E, int Etot, int H) {
  int idx = blockIdx.x * blockDim.x + threadIdx.x;
  if (idx >= Etot * H) return;
  int e = idx / H, hh = idx - e * H;
  int s, d;
  if (e < E) { s = src[e]; d = dst[e]; } else { s = d = e - E; }
  float v = as[s * H + hh] + ad[d * H + hh];
  v = v > 0.f ? v : LRELU_SLOPE * v;
  ebuf[idx] = v;
  atomicMax(&mkey[d * H + hh], f32_key(v));
}

// Pass B: w = exp(e - m[dst]); store w; atomicAdd denom[dst]
__global__ void gat_edge_expsum(const int* __restrict__ dst,
                                float* __restrict__ ebuf,
                                const unsigned* __restrict__ mkey,
                                float* __restrict__ denom,
                                int E, int Etot, int H) {
  int idx = blockIdx.x * blockDim.x + threadIdx.x;
  if (idx >= Etot * H) return;
  int e = idx / H, hh = idx - e * H;
  int d = (e < E) ? dst[e] : (e - E);
  float m = key_f32(mkey[d * H + hh]);
  float w = expf(ebuf[idx] - m);
  ebuf[idx] = w;
  atomicAdd(&denom[d * H + hh], w);
}

// Pass C: acc[dst] += h[src] * (w / denom[dst]), float4 gather + 4 f32 atomics
__global__ void gat_edge_scatter(const int* __restrict__ src, const int* __restrict__ dst,
                                 const float* __restrict__ ebuf, const float* __restrict__ denom,
                                 const float* __restrict__ h, float* __restrict__ acc,
                                 int E, int Etot, int H, int F) {
  const int HF4 = (H * F) >> 2;
  unsigned idx = blockIdx.x * blockDim.x + threadIdx.x;
  if (idx >= (unsigned)Etot * (unsigned)HF4) return;
  int e = idx / HF4, q = idx - e * HF4;
  int s, d;
  if (e < E) { s = src[e]; d = dst[e]; } else { s = d = e - E; }
  int hh = (q * 4) / F;
  float alpha = ebuf[e * H + hh] / denom[d * H + hh];
  float4 hv = *(const float4*)(h + (size_t)s * H * F + q * 4);
  float* ap = acc + (size_t)d * H * F + q * 4;
  atomicAdd(ap + 0, hv.x * alpha);
  atomicAdd(ap + 1, hv.y * alpha);
  atomicAdd(ap + 2, hv.z * alpha);
  atomicAdd(ap + 3, hv.w * alpha);
}

// out = elu(acc + bias), in place
__global__ void gat_bias_elu(float* __restrict__ acc, const float* __restrict__ bias,
                             int N, int C) {
  int idx = blockIdx.x * blockDim.x + threadIdx.x;
  if (idx >= N * C) return;
  float v = acc[idx] + bias[idx % C];
  acc[idx] = v > 0.f ? v : (expf(v) - 1.f);
}

// MLP head: relu(h@fc1 + b1) @ fc2 + b2  -> out[N]
__global__ __launch_bounds__(256) void gat_mlp(const float* __restrict__ h,
                                               const float* __restrict__ fc1w,
                                               const float* __restrict__ fc1b,
                                               const float* __restrict__ fc2w,
                                               const float* __restrict__ fc2b,
                                               float* __restrict__ out, int N) {
  __shared__ float w1[64 * 32];
  __shared__ float b1[32];
  __shared__ float w2[32];
  __shared__ float b2s;
  for (int i = threadIdx.x; i < 64 * 32; i += 256) w1[i] = fc1w[i];
  if (threadIdx.x < 32) { b1[threadIdx.x] = fc1b[threadIdx.x]; w2[threadIdx.x] = fc2w[threadIdx.x]; }
  if (threadIdx.x == 0) b2s = fc2b[0];
  __syncthreads();
  int n = blockIdx.x * 256 + threadIdx.x;
  if (n >= N) return;
  float x[64];
  const float4* hp = (const float4*)(h + (size_t)n * 64);
#pragma unroll
  for (int i = 0; i < 16; ++i) {
    float4 v = hp[i];
    x[4 * i + 0] = v.x; x[4 * i + 1] = v.y; x[4 * i + 2] = v.z; x[4 * i + 3] = v.w;
  }
  float o = 0.f;
#pragma unroll 4
  for (int j = 0; j < 32; ++j) {
    float t = b1[j];
    for (int f = 0; f < 64; ++f) t += x[f] * w1[f * 32 + j];
    t = t > 0.f ? t : 0.f;
    o += t * w2[j];
  }
  out[n] = o + b2s;
}

// ---------------------------------------------------------------------------
extern "C" void kernel_launch(void* const* d_in, const int* in_sizes, int n_in,
                              void* d_out, int out_size, void* d_ws, size_t ws_size,
                              hipStream_t stream) {
  (void)n_in; (void)out_size;
  const float* x    = (const float*)d_in[0];
  const int*   ei   = (const int*)d_in[1];
  const float* W1   = (const float*)d_in[2];
  const float* aS1  = (const float*)d_in[3];
  const float* aD1  = (const float*)d_in[4];
  const float* b1   = (const float*)d_in[5];
  const float* W2   = (const float*)d_in[6];
  const float* aS2  = (const float*)d_in[7];
  const float* aD2  = (const float*)d_in[8];
  const float* b2   = (const float*)d_in[9];
  const float* fc1w = (const float*)d_in[10];
  const float* fc1b = (const float*)d_in[11];
  const float* fc2w = (const float*)d_in[12];
  const float* fc2b = (const float*)d_in[13];

  const int N    = in_sizes[0] / 128;
  const int E    = in_sizes[1] / 2;
  const int Etot = E + N;                 // with self-loops
  const int* src = ei;
  const int* dst = ei + E;

  char* ws = (char*)d_ws;
  size_t off = 0;
  auto alloc = [&](size_t bytes) -> char* {
    char* p = ws + off;
    off = (off + bytes + 255) & ~(size_t)255;
    return p;
  };
  float*    h1    = (float*)alloc((size_t)N * 128 * 4);
  float*    acc1  = (float*)alloc((size_t)N * 128 * 4);
  float*    h2    = (float*)alloc((size_t)N * 64 * 4);
  float*    acc2  = (float*)alloc((size_t)N * 64 * 4);
  float*    as1   = (float*)alloc((size_t)N * 2 * 4);
  float*    ad1   = (float*)alloc((size_t)N * 2 * 4);
  float*    as2   = (float*)alloc((size_t)N * 4);
  float*    ad2   = (float*)alloc((size_t)N * 4);
  unsigned* mkey1 = (unsigned*)alloc((size_t)N * 2 * 4);
  float*    den1  = (float*)alloc((size_t)N * 2 * 4);
  unsigned* mkey2 = (unsigned*)alloc((size_t)N * 4);
  float*    den2  = (float*)alloc((size_t)N * 4);
  float*    e1    = (float*)alloc((size_t)Etot * 2 * 4);
  float*    e2    = (float*)alloc((size_t)Etot * 4);
  unsigned short* wp1 = (unsigned short*)alloc(128 * 128 * 2);
  unsigned short* wp2 = (unsigned short*)alloc(128 * 64 * 2);
  if (off > ws_size) return;  // insufficient scratch

  const int B = 256;
  auto blocks = [&](long long t) { return (unsigned)((t + B - 1) / B); };

  // Zero-init accumulators / segment-max keys / denominators (every call).
  hipMemsetAsync(acc1,  0, (size_t)N * 128 * 4, stream);
  hipMemsetAsync(acc2,  0, (size_t)N * 64 * 4,  stream);
  hipMemsetAsync(mkey1, 0, (size_t)N * 2 * 4,   stream);
  hipMemsetAsync(den1,  0, (size_t)N * 2 * 4,   stream);
  hipMemsetAsync(mkey2, 0, (size_t)N * 4,       stream);
  hipMemsetAsync(den2,  0, (size_t)N * 4,       stream);

  // Pre-swizzle weights into WMMA B fragments.
  gat_pack_w<<<blocks(4 * 8 * 512), B, 0, stream>>>(W1, wp1, 128, 128);
  gat_pack_w<<<blocks(4 * 4 * 512), B, 0, stream>>>(W2, wp2, 128, 64);

  // ---- Layer 1 (H=2, F=64) ----
  gat_gemm_wmma<8><<<(N + 127) / 128, B, 0, stream>>>(x, (const bf16x16*)wp1, h1, N);
  gat_attn_alpha<<<blocks((long long)N * 2), B, 0, stream>>>(h1, aS1, aD1, as1, ad1, N, 2, 64);
  gat_edge_logit_max<<<blocks((long long)Etot * 2), B, 0, stream>>>(src, dst, as1, ad1, e1, mkey1, E, Etot, 2);
  gat_edge_expsum<<<blocks((long long)Etot * 2), B, 0, stream>>>(dst, e1, mkey1, den1, E, Etot, 2);
  gat_edge_scatter<<<blocks((long long)Etot * 32), B, 0, stream>>>(src, dst, e1, den1, h1, acc1, E, Etot, 2, 64);
  gat_bias_elu<<<blocks((long long)N * 128), B, 0, stream>>>(acc1, b1, N, 128);

  // ---- Layer 2 (H=1, F=64) ----
  gat_gemm_wmma<4><<<(N + 127) / 128, B, 0, stream>>>(acc1, (const bf16x16*)wp2, h2, N);
  gat_attn_alpha<<<blocks((long long)N), B, 0, stream>>>(h2, aS2, aD2, as2, ad2, N, 1, 64);
  gat_edge_logit_max<<<blocks((long long)Etot), B, 0, stream>>>(src, dst, as2, ad2, e2, mkey2, E, Etot, 1);
  gat_edge_expsum<<<blocks((long long)Etot), B, 0, stream>>>(dst, e2, mkey2, den2, E, Etot, 1);
  gat_edge_scatter<<<blocks((long long)Etot * 16), B, 0, stream>>>(src, dst, e2, den2, h2, acc2, E, Etot, 1, 64);
  gat_bias_elu<<<blocks((long long)N * 64), B, 0, stream>>>(acc2, b2, N, 64);

  // ---- MLP head ----
  gat_mlp<<<blocks((long long)N), B, 0, stream>>>(acc2, fc1w, fc1b, fc2w, fc2b, (float*)d_out, N);
}
